// MultiHeadAttention_61658550501705
// MI455X (gfx1250) — matrix operand order repaired and checked by reading, not compiled, not run on hardware
//
#include <hip/hip_runtime.h>
#include <hip/hip_bf16.h>

// ---------------------------------------------------------------------------
// MultiHeadAttention forward for MI455X (gfx1250, wave32, WMMA).
//   B=2, S=2048, d_model=1024, H=16, depth=64.
// Compute-bound (~69 GFLOP vs ~80MB HBM) -> v_wmma_f32_16x16x32_f16 everywhere
// (f16 operands, f32 accumulation). Flash-attention middle stage keeps the
// 2048x2048 logits on-chip. Data movement uses the CDNA5 paths:
//   - TDM tensor_load_to_lds (inline asm, 2-group D#) for the K tile with
//     hardware LDS padding to stride 72 halfs
//   - global_load_async_to_lds_b128 (ASYNCcnt) for Q/mask/A-tile staging
//   - global_prefetch_b8 for the next GEMM K-step
// ---------------------------------------------------------------------------

typedef __attribute__((ext_vector_type(16))) _Float16 v16h;
typedef __attribute__((ext_vector_type(8)))  _Float16 v8h;
typedef __attribute__((ext_vector_type(8)))  float    v8f;
typedef __attribute__((ext_vector_type(4)))  unsigned v4u;
typedef __attribute__((ext_vector_type(8)))  int      v8i;

#define D_MODEL 1024
#define NUM_HEADS 16
#define DEPTH 64
#define BATCH 2
#define SEQ 2048
#define NEG_INF_F (-1.0e9f)

__device__ __forceinline__ v16h pack16(v8h lo, v8h hi) {
  return __builtin_shufflevector(lo, hi, 0,1,2,3,4,5,6,7,8,9,10,11,12,13,14,15);
}

__device__ __forceinline__ v8f wmma16(v16h a, v16h b, v8f c) {
  // v_wmma_f32_16x16x32_f16  D = A(16x32 f16) * B(32x16 f16) + C(16x16 f32)
  return __builtin_amdgcn_wmma_f32_16x16x32_f16(false, a, false, b,
                                                (short)0, c, false, false);
}

// A-fragment (16x32 f16): lane<16 -> row=lane, K = {cb..cb+7, cb+16..cb+23} cb=0
//                         lane>=16 -> same row, cb=8. Row-major LDS, contiguous.
__device__ __forceinline__ v16h frag_a(const _Float16* row_base, int lane, int koff) {
  const int cb = koff + ((lane >> 4) << 3);
  return pack16(*(const v8h*)(row_base + cb), *(const v8h*)(row_base + cb + 16));
}

// B-fragment (32x16 f16): lane holds column N=lane&15; lane<16 -> K = {kb..kb+7, kb+16..kb+23} kb=koff,
// lane>=16 -> kb=koff+8.  N-major LDS tile.
__device__ __forceinline__ v16h frag_b(const _Float16* col_base, int lane, int koff) {
  const int kb = koff + ((lane >> 4) << 3);
  return pack16(*(const v8h*)(col_base + kb), *(const v8h*)(col_base + kb + 16));
}

// Async 16B global->LDS copy (ASYNCcnt). Per ISA 10.2, the low 32 bits of a
// generic LDS pointer are the LDS byte address.
__device__ __forceinline__ void async_copy_b128(void* lds_dst, const void* gsrc) {
  asm volatile("global_load_async_to_lds_b128 %0, %1, off"
               :: "v"((unsigned)(unsigned long long)lds_dst), "v"(gsrc)
               : "memory");
}
__device__ __forceinline__ void wait_async() {
  asm volatile("s_wait_asynccnt 0x0" ::: "memory");
}

// TDM DMA: 2D tensor tile -> LDS, descriptor groups 0 (4 SGPRs) and 1
// (8 SGPRs); groups 2/3 omitted (NULL) -> 2D form per ISA 7.3.
__device__ __forceinline__ void tdm_load_2d(v4u g0, v8i g1) {
  asm volatile("tensor_load_to_lds %0, %1" :: "s"(g0), "s"(g1) : "memory");
}

// ---------------------------------------------------------------------------
// Kernel 1: Y = X @ W (W in {Wq,Wk,Wv}), f32 in -> f16 out in [B,H,S,64] layout.
// Block: 256 thr (8 waves), tile 128(M) x 64(N), K-step 32.
// ---------------------------------------------------------------------------
__global__ __launch_bounds__(256)
void proj_qkv_kernel(const float* __restrict__ X,
                     const float* __restrict__ Wq,
                     const float* __restrict__ Wk,
                     const float* __restrict__ Wv,
                     _Float16* __restrict__ qf,
                     _Float16* __restrict__ kf,
                     _Float16* __restrict__ vf) {
  const float* W = (blockIdx.z == 0) ? Wq : (blockIdx.z == 1) ? Wk : Wv;
  _Float16*    O = (blockIdx.z == 0) ? qf : (blockIdx.z == 1) ? kf : vf;

  __shared__ __align__(16) _Float16 As[128][40];  // row-major (M x K)
  __shared__ __align__(16) _Float16 Bs[64][40];   // n-major  (N x K)

  const int tid  = threadIdx.x;
  const int lane = tid & 31;
  const int wv   = tid >> 5;
  const int wm   = (wv >> 1) * 32;
  const int wn   = (wv & 1) * 32;
  const int m0   = blockIdx.y * 128;
  const int n0   = blockIdx.x * 64;

  const v8f zero = {};
  v8f acc[2][2];
  acc[0][0] = zero; acc[0][1] = zero; acc[1][0] = zero; acc[1][1] = zero;

  const int arow = tid >> 1;
  const int aseg = (tid & 1) * 16;
  const int bn   = tid & 63;
  const int bks  = (tid >> 6) * 8;

  for (int k0 = 0; k0 < D_MODEL; k0 += 32) {
    __syncthreads();
    {  // A tile 128x32 : f32 -> f16
      const float4* src = (const float4*)(X + (size_t)(m0 + arow) * D_MODEL + k0 + aseg);
      if (k0 + 32 < D_MODEL)
        __builtin_prefetch(X + (size_t)(m0 + arow) * D_MODEL + k0 + 32 + aseg, 0, 0);
#pragma unroll
      for (int i = 0; i < 4; ++i) {
        float4 v = src[i];
        As[arow][aseg + i*4 + 0] = (_Float16)v.x;
        As[arow][aseg + i*4 + 1] = (_Float16)v.y;
        As[arow][aseg + i*4 + 2] = (_Float16)v.z;
        As[arow][aseg + i*4 + 3] = (_Float16)v.w;
      }
    }
    {  // B tile 32x64 -> Bs[n][k] (transposed), f32 -> f16
#pragma unroll
      for (int j = 0; j < 8; ++j) {
        float w = W[(size_t)(k0 + bks + j) * D_MODEL + n0 + bn];
        Bs[bn][bks + j] = (_Float16)w;
      }
    }
    __syncthreads();

    v16h af[2], bf[2];
#pragma unroll
    for (int mt = 0; mt < 2; ++mt)
      af[mt] = frag_a(&As[wm + mt*16 + (lane & 15)][0], lane, 0);
#pragma unroll
    for (int nt = 0; nt < 2; ++nt)
      bf[nt] = frag_b(&Bs[wn + nt*16 + (lane & 15)][0], lane, 0);
#pragma unroll
    for (int mt = 0; mt < 2; ++mt)
#pragma unroll
      for (int nt = 0; nt < 2; ++nt)
        acc[mt][nt] = wmma16(af[mt], bf[nt], acc[mt][nt]);
  }

  // Epilogue: C-layout lane = (col = lane&15, rows r + 8*(lane>=16)).
  const int rbase = (lane >> 4) << 3;
#pragma unroll
  for (int mt = 0; mt < 2; ++mt) {
#pragma unroll
    for (int nt = 0; nt < 2; ++nt) {
      const int col = n0 + wn + nt*16 + (lane & 15);
      const int h = col >> 6, d = col & 63;
#pragma unroll
      for (int r = 0; r < 8; ++r) {
        const int m = m0 + wm + mt*16 + r + rbase;
        const int b = m >> 11, s = m & (SEQ - 1);
        O[((size_t)(b * NUM_HEADS + h) * SEQ + s) * DEPTH + d] =
            (_Float16)acc[mt][nt][r];
      }
    }
  }
}

// ---------------------------------------------------------------------------
// Kernel 2: flash attention per (64-query tile, b*H+h).
// Block: 128 thr (4 waves); each wave owns 16 query rows, loops 64-key tiles.
// K tile via TDM (hardware pad -> stride 72 halfs); Q/mask via async-to-LDS.
// ---------------------------------------------------------------------------
__global__ __launch_bounds__(128)
void flash_attn_kernel(const _Float16* __restrict__ qf,
                       const _Float16* __restrict__ kf,
                       const _Float16* __restrict__ vf,
                       const int* __restrict__ mask,
                       _Float16* __restrict__ of) {
  __shared__ __align__(16) _Float16 Qs[64][72];       // [q][d]   row-major
  __shared__ __align__(16) _Float16 Ks[64][72];       // [key][d] == B n-major
  __shared__ __align__(16) _Float16 Vs[64][72];       // [d][key] (transposed)
  __shared__ __align__(16) _Float16 Ps[4][16][72];    // per-wave P (A-layout src)
  __shared__ __align__(16) int      Msi[64][68];      // mask tile (raw i32)

  const int tid  = threadIdx.x;
  const int lane = tid & 31;
  const int w    = tid >> 5;            // 0..3
  const int bh   = blockIdx.y;          // b*16 + h
  const int b    = bh >> 4;
  const int h    = bh & 15;
  const int q0   = blockIdx.x * 64;

  const _Float16* qbase = qf + (size_t)bh * SEQ * DEPTH;
  const _Float16* kbase = kf + (size_t)bh * SEQ * DEPTH;
  const _Float16* vbase = vf + (size_t)bh * SEQ * DEPTH;
  const int*      mbase = mask + (size_t)b * SEQ * SEQ;

  const int lrow = tid >> 1;            // 0..63
  const int lseg = (tid & 1) * 32;      // 0 / 32

  {  // stage Q tile (row-major) via async global->LDS
    const _Float16* src = qbase + (size_t)(q0 + lrow) * DEPTH + lseg;
#pragma unroll
    for (int i = 0; i < 4; ++i)
      async_copy_b128(&Qs[lrow][lseg + i*8], src + i*8);
  }
  wait_async();
  __syncthreads();

  v16h aq[2];
  {
    const _Float16* p = &Qs[w*16 + (lane & 15)][0];
    aq[0] = frag_a(p, lane, 0);
    aq[1] = frag_a(p, lane, 32);
  }

  const v8f zero = {};
  v8f o[4];
  o[0] = zero; o[1] = zero; o[2] = zero; o[3] = zero;
  float m_run[8], l_run[8];
#pragma unroll
  for (int r = 0; r < 8; ++r) { m_run[r] = -3.0e38f; l_run[r] = 0.0f; }

  const float scale = 0.125f;           // 1/sqrt(64)
  const int rbase = (lane >> 4) << 3;
  const int ncol  = lane & 15;

  for (int kt = 0; kt < SEQ; kt += 64) {
    __syncthreads();  // previous iteration's LDS reads complete

    if (w == 0) {
      // TDM: 64x64 f16 tile from kbase+kt*64, row stride 64 elems, into Ks
      // with hardware padding: after every 32 DWORDs (one 128B row) insert
      // 4 DWORDs (16B) -> LDS row stride 144B = 72 halfs.
      const unsigned ldsK = (unsigned)(unsigned long long)&Ks[0][0];
      const unsigned long long ga =
          (unsigned long long)(const void*)(kbase + (size_t)kt * DEPTH);
      v4u g0 = { 1u,                                   // count=1 (valid D#)
                 ldsK,                                 // lds_addr
                 (unsigned)(ga & 0xFFFFFFFFu),         // global_addr[31:0]
                 (unsigned)((ga >> 32) & 0x1FFFFFFu) | 0x80000000u }; // +type=2
      v8i g1 = { (int)0x07110000,   // data_size=2B, pad_en, intv=32dw, amt=4dw
                 (int)(64u << 16),  // tensor_dim0 = 64
                 (int)(64u << 16),  // tensor_dim1 = 64
                 (int)(64u << 16),  // tile_dim0 = 64
                 64,                // tile_dim1 = 64 (tile_dim2 = 0)
                 64,                // tensor_dim0_stride = 64
                 0, 0 };
      tdm_load_2d(g0, g1);
      __builtin_amdgcn_s_wait_tensorcnt(0);
    }
    {  // V tile transposed in registers: Vs[d][key]
      const v8h* src = (const v8h*)(vbase + (size_t)(kt + lrow) * DEPTH + lseg);
#pragma unroll
      for (int i = 0; i < 4; ++i) {
        v8h t = src[i];
#pragma unroll
        for (int j = 0; j < 8; ++j) Vs[lseg + i*8 + j][lrow] = t[j];
      }
    }
    {  // mask tile via async global->LDS (convert i32->f32 at consumption)
      const int* src = mbase + (size_t)(q0 + lrow) * SEQ + kt + lseg;
#pragma unroll
      for (int i = 0; i < 8; ++i)
        async_copy_b128(&Msi[lrow][lseg + i*4], src + i*4);
    }
    wait_async();
    __syncthreads();

    // logits: 4 key sub-tiles x 2 depth chunks
    v8f cl[4];
    cl[0] = zero; cl[1] = zero; cl[2] = zero; cl[3] = zero;
#pragma unroll
    for (int nsub = 0; nsub < 4; ++nsub) {
      const _Float16* p = &Ks[nsub*16 + (lane & 15)][0];
#pragma unroll
      for (int c = 0; c < 2; ++c)
        cl[nsub] = wmma16(aq[c], frag_b(p, lane, c*32), cl[nsub]);
    }

    // scale + mask, then online softmax (row = query, reduce over 16 lanes)
    float mnew[8];
#pragma unroll
    for (int r = 0; r < 8; ++r) mnew[r] = m_run[r];
#pragma unroll
    for (int nsub = 0; nsub < 4; ++nsub) {
#pragma unroll
      for (int r = 0; r < 8; ++r) {
        float x = cl[nsub][r] * scale +
                  (float)Msi[w*16 + r + rbase][nsub*16 + ncol] * NEG_INF_F;
        cl[nsub][r] = x;
        mnew[r] = fmaxf(mnew[r], x);
      }
    }
#pragma unroll
    for (int r = 0; r < 8; ++r)
#pragma unroll
      for (int off = 1; off < 16; off <<= 1)
        mnew[r] = fmaxf(mnew[r], __shfl_xor(mnew[r], off, 32));

    float rsum[8];
#pragma unroll
    for (int r = 0; r < 8; ++r) rsum[r] = 0.0f;
#pragma unroll
    for (int nsub = 0; nsub < 4; ++nsub) {
#pragma unroll
      for (int r = 0; r < 8; ++r) {
        float p = __expf(cl[nsub][r] - mnew[r]);
        rsum[r] += p;
        Ps[w][r + rbase][nsub*16 + ncol] = (_Float16)p;   // C-layout -> LDS
      }
    }
#pragma unroll
    for (int r = 0; r < 8; ++r) {
#pragma unroll
      for (int off = 1; off < 16; off <<= 1)
        rsum[r] += __shfl_xor(rsum[r], off, 32);
      float alpha = __expf(m_run[r] - mnew[r]);
      l_run[r] = l_run[r] * alpha + rsum[r];
      m_run[r] = mnew[r];
#pragma unroll
      for (int dsub = 0; dsub < 4; ++dsub) o[dsub][r] *= alpha;
    }

    __syncthreads();  // P visible cross-lane; Vs ready

    // O += P @ V : A = P (re-read in A-layout), B = V (Vs is d-major)
#pragma unroll
    for (int c = 0; c < 2; ++c) {
      v16h ap = frag_a(&Ps[w][lane & 15][0], lane, c*32);
#pragma unroll
      for (int dsub = 0; dsub < 4; ++dsub) {
        v16h bv = frag_b(&Vs[dsub*16 + (lane & 15)][0], lane, c*32);
        o[dsub] = wmma16(ap, bv, o[dsub]);
      }
    }
  }

  // finalize: divide by softmax denominator, write [B,S,H*64] f16
#pragma unroll
  for (int dsub = 0; dsub < 4; ++dsub) {
    const int col = h*64 + dsub*16 + (lane & 15);
#pragma unroll
    for (int r = 0; r < 8; ++r) {
      const int q = q0 + w*16 + r + rbase;
      of[((size_t)b * SEQ + q) * D_MODEL + col] =
          (_Float16)(o[dsub][r] / l_run[r]);
    }
  }
}

// ---------------------------------------------------------------------------
// Kernel 3: out(f32) = attn(f16) @ Wo(f32->f16). Same tiling as kernel 1;
// f16 A tile staged with async global->LDS.
// ---------------------------------------------------------------------------
__global__ __launch_bounds__(256)
void proj_out_kernel(const _Float16* __restrict__ A,
                     const float* __restrict__ Wo,
                     float* __restrict__ out) {
  __shared__ __align__(16) _Float16 As[128][40];
  __shared__ __align__(16) _Float16 Bs[64][40];

  const int tid  = threadIdx.x;
  const int lane = tid & 31;
  const int wv   = tid >> 5;
  const int wm   = (wv >> 1) * 32;
  const int wn   = (wv & 1) * 32;
  const int m0   = blockIdx.y * 128;
  const int n0   = blockIdx.x * 64;

  const v8f zero = {};
  v8f acc[2][2];
  acc[0][0] = zero; acc[0][1] = zero; acc[1][0] = zero; acc[1][1] = zero;

  const int arow = tid >> 1;
  const int aseg = (tid & 1) * 16;
  const int bn   = tid & 63;
  const int bks  = (tid >> 6) * 8;

  for (int k0 = 0; k0 < D_MODEL; k0 += 32) {
    __syncthreads();
    {  // A tile: f16 straight copy via async global->LDS
      const _Float16* src = A + (size_t)(m0 + arow) * D_MODEL + k0 + aseg;
      if (k0 + 32 < D_MODEL)
        __builtin_prefetch(src + 32, 0, 0);
      async_copy_b128(&As[arow][aseg],     src);
      async_copy_b128(&As[arow][aseg + 8], src + 8);
    }
    {
#pragma unroll
      for (int j = 0; j < 8; ++j) {
        float w = Wo[(size_t)(k0 + bks + j) * D_MODEL + n0 + bn];
        Bs[bn][bks + j] = (_Float16)w;
      }
    }
    wait_async();
    __syncthreads();

    v16h af[2], bf[2];
#pragma unroll
    for (int mt = 0; mt < 2; ++mt)
      af[mt] = frag_a(&As[wm + mt*16 + (lane & 15)][0], lane, 0);
#pragma unroll
    for (int nt = 0; nt < 2; ++nt)
      bf[nt] = frag_b(&Bs[wn + nt*16 + (lane & 15)][0], lane, 0);
#pragma unroll
    for (int mt = 0; mt < 2; ++mt)
#pragma unroll
      for (int nt = 0; nt < 2; ++nt)
        acc[mt][nt] = wmma16(af[mt], bf[nt], acc[mt][nt]);
  }

  const int rbase = (lane >> 4) << 3;
#pragma unroll
  for (int mt = 0; mt < 2; ++mt) {
#pragma unroll
    for (int nt = 0; nt < 2; ++nt) {
      const int col = n0 + wn + nt*16 + (lane & 15);
#pragma unroll
      for (int r = 0; r < 8; ++r) {
        const int m = m0 + wm + mt*16 + r + rbase;
        out[(size_t)m * D_MODEL + col] = acc[mt][nt][r];
      }
    }
  }
}

// ---------------------------------------------------------------------------
extern "C" void kernel_launch(void* const* d_in, const int* in_sizes, int n_in,
                              void* d_out, int out_size, void* d_ws, size_t ws_size,
                              hipStream_t stream) {
  const float* X    = (const float*)d_in[0];   // [B,S,1024] f32
  const int*   mask = (const int*)d_in[1];     // [B,1,S,S]  i32
  const float* Wq   = (const float*)d_in[2];
  const float* Wk   = (const float*)d_in[3];
  const float* Wv   = (const float*)d_in[4];
  const float* Wo   = (const float*)d_in[5];
  float* out = (float*)d_out;                  // [B,S,1024] f32

  // Workspace: 4 x 8MB f16 intermediates (32 MB total)
  _Float16* ws = (_Float16*)d_ws;
  const size_t HSZ = (size_t)BATCH * NUM_HEADS * SEQ * DEPTH;  // 4,194,304
  _Float16* qf = ws;
  _Float16* kf = ws + HSZ;
  _Float16* vf = ws + 2 * HSZ;
  _Float16* of = ws + 3 * HSZ;                 // [B,S,1024] f16

  // 1) Q/K/V projections (grid.z selects weight), M=4096 N=1024
  proj_qkv_kernel<<<dim3(D_MODEL / 64, (BATCH * SEQ) / 128, 3), 256, 0, stream>>>(
      X, Wq, Wk, Wv, qf, kf, vf);

  // 2) flash attention: 32 query tiles x (B*H = 32)
  flash_attn_kernel<<<dim3(SEQ / 64, BATCH * NUM_HEADS), 128, 0, stream>>>(
      qf, kf, vf, mask, of);

  // 3) output projection
  proj_out_kernel<<<dim3(D_MODEL / 64, (BATCH * SEQ) / 128), 256, 0, stream>>>(
      of, Wo, out);
}
